// Policy_90477781058033
// MI455X (gfx1250) — compile-verified, matrix operand order checked by reading, not run on hardware
//
#include <hip/hip_runtime.h>
#include <hip/hip_bf16.h>

// Problem constants (match reference)
#define T_STEPS 8192
#define NS_DIM  512
#define NH_DIM  1024
#define NO_DIM  512
#define G4      (4 * NH_DIM)   // 4096 gate rows per layer
#define REC_WGS 128            // persistent workgroups (must all be co-resident)

typedef __attribute__((ext_vector_type(16))) __bf16       v16bf;
typedef __attribute__((ext_vector_type(8)))  float        v8f;
typedef __attribute__((ext_vector_type(8)))  unsigned int v8u;

// ---------------------------------------------------------------------------
// Helpers
// ---------------------------------------------------------------------------
__device__ __forceinline__ unsigned short f32_to_bf16_rne(float f) {
  unsigned u = __float_as_uint(f);
  return (unsigned short)((u + 0x7FFFu + ((u >> 16) & 1u)) >> 16);
}
__device__ __forceinline__ float sigmoidf_(float x) {
  return 1.0f / (1.0f + __expf(-x));
}

// ---------------------------------------------------------------------------
// Elementwise f32 -> bf16 conversion
// ---------------------------------------------------------------------------
__global__ void conv_bf16_kernel(const float* __restrict__ in,
                                 unsigned short* __restrict__ out, int n) {
  int i = blockIdx.x * blockDim.x + threadIdx.x;
  if (i < n) out[i] = f32_to_bf16_rne(in[i]);
}

// Concatenate layer-1 weights row-wise: out[row][0:NH]=w_ih1, [NH:2NH]=w_hh1 (bf16)
__global__ void concat_w1_kernel(const float* __restrict__ wih1,
                                 const float* __restrict__ whh1,
                                 unsigned short* __restrict__ out) {
  int i = blockIdx.x * blockDim.x + threadIdx.x;     // over G4 * 2NH
  if (i >= G4 * 2 * NH_DIM) return;
  int row = i >> 11;            // /2048
  int k   = i & 2047;
  float f = (k < NH_DIM) ? wih1[(size_t)row * NH_DIM + k]
                         : whh1[(size_t)row * NH_DIM + (k - NH_DIM)];
  out[i] = f32_to_bf16_rne(f);
}

// Per-launch init: combined biases, zeroed h ping-pong buffers, barrier counters
__global__ void init_kernel(const float* __restrict__ bih0, const float* __restrict__ bhh0,
                            const float* __restrict__ bih1, const float* __restrict__ bhh1,
                            float* __restrict__ b0, float* __restrict__ b1,
                            float* __restrict__ h0buf, float* __restrict__ h1buf,
                            unsigned* __restrict__ syncp) {
  int i = blockIdx.x * blockDim.x + threadIdx.x;
  if (i < G4) { b0[i] = bih0[i] + bhh0[i]; b1[i] = bih1[i] + bhh1[i]; }
  if (i < 2 * NH_DIM) { h0buf[i] = 0.0f; h1buf[i] = 0.0f; }
  if (i < 2) syncp[i] = 0u;
}

// ---------------------------------------------------------------------------
// WMMA bf16 GEMM:  C[M,N] = A[M,K] * Bt[N,K]^T + bias[N]
// One wave computes one 16x16 C tile; K-loop in chunks of 32 using
// v_wmma_f32_16x16x32_bf16.  Fragment layouts per CDNA5 ISA 7.12.2:
//   16-bit A (16x32): lane L holds M=L%16; VGPR v packs K = f(v, L/16), f+1
//   with f(v,h) = (v/4)*16 + (v%4)*2 + h*8.  B (32x16) mirrors with N=L%16,
//   so for Bt row-major both fragments are consecutive-K dword loads.
//   C: lane L -> N=L%16, VGPR v -> M = v + 8*(L/16).
// ---------------------------------------------------------------------------
__global__ void __launch_bounds__(256) wmma_gemm_bf16(
    const unsigned short* __restrict__ A,   // [M,K] bf16 row-major
    const unsigned short* __restrict__ Bt,  // [N,K] bf16 row-major (B transposed)
    const float* __restrict__ bias,         // [N]
    float* __restrict__ C,                  // [M,N] f32
    int M, int N, int K) {
  const int wave    = blockIdx.x * (blockDim.x >> 5) + (threadIdx.x >> 5);
  const int tiles_n = N >> 4;
  const int total   = (M >> 4) * tiles_n;
  if (wave >= total) return;
  const int tm   = wave / tiles_n;
  const int tn   = wave % tiles_n;
  const int lane = threadIdx.x & 31;
  const int half = lane >> 4;
  const int l16  = lane & 15;

  int kb[8];
#pragma unroll
  for (int v = 0; v < 8; ++v)
    kb[v] = ((v >> 2) << 4) + ((v & 3) << 1) + (half << 3);

  const unsigned* Arow =
      reinterpret_cast<const unsigned*>(A + (size_t)(tm * 16 + l16) * K);
  const unsigned* Brow =
      reinterpret_cast<const unsigned*>(Bt + (size_t)(tn * 16 + l16) * K);

  v8f c = {0.f, 0.f, 0.f, 0.f, 0.f, 0.f, 0.f, 0.f};
  for (int kc = 0; kc < K; kc += 32) {
    v8u au, bu;
#pragma unroll
    for (int v = 0; v < 8; ++v) {
      au[v] = Arow[(kc + kb[v]) >> 1];
      bu[v] = Brow[(kc + kb[v]) >> 1];
    }
    c = __builtin_amdgcn_wmma_f32_16x16x32_bf16(
        false, __builtin_bit_cast(v16bf, au),
        false, __builtin_bit_cast(v16bf, bu),
        (short)0, c, false, false);
  }

  const int ncol = tn * 16 + l16;
  const float bv = bias ? bias[ncol] : 0.0f;
#pragma unroll
  for (int v = 0; v < 8; ++v) {
    const int mrow = tm * 16 + v + (half << 3);
    C[(size_t)mrow * N + ncol] = c[v] + bv;
  }
}

// ---------------------------------------------------------------------------
// Persistent 2-layer LSTM recurrence.  128 WGs x 256 threads, all resident.
// Mapping: wave w of WG b owns cell = b*8+w of both layers; within the wave,
// lanes [8g..8g+7] compute gate g's dot product (i,f,g,o).  Cell state c in
// registers; h vectors ping-pong in global (L2).  One device-wide barrier per
// timestep (between layer-0 writes and layer-1 reads); correctness of the
// single barrier follows from ping-pong parity + per-WG program order.
// L2 traffic: 24 MB bf16 weights per step, resident in the 192 MB L2.
// ---------------------------------------------------------------------------
__global__ void __launch_bounds__(256, 1) lstm_persistent(
    const unsigned short* __restrict__ Whh0,   // [4096,1024] bf16
    const unsigned short* __restrict__ W1cat,  // [4096,2048] bf16 ([w_ih1|w_hh1])
    const float* __restrict__ Z0,              // [T,4096] x-projection + b0
    const float* __restrict__ b1,              // [4096]
    float* __restrict__ h0buf,                 // [2,1024] ping-pong
    float* __restrict__ h1buf,                 // [2,1024] ping-pong
    float* __restrict__ hs,                    // [T,1024] layer-1 outputs
    unsigned* __restrict__ syncp) {            // [0]=arrive count, [1]=generation
  __shared__ float sh[2 * NH_DIM];
  const int tid  = threadIdx.x;
  const int lane = tid & 31;
  const int grp  = lane >> 3;   // gate index 0..3 (i,f,g,o)
  const int gl   = lane & 7;    // lane within gate-dot group
  const int cell = blockIdx.x * 8 + (tid >> 5);
  const int row  = grp * NH_DIM + cell;
  const unsigned short* w0 = Whh0 + (size_t)row * NH_DIM;
  const unsigned short* w1 = W1cat + (size_t)row * (2 * NH_DIM);
  volatile unsigned* vgen = syncp + 1;

  float c0 = 0.0f, c1 = 0.0f;

  for (int t = 0; t < T_STEPS; ++t) {
    const int p = t & 1, rp = p ^ 1;

    // ---- layer 0: gates = Z0[t] + W_hh0 @ h0_prev -------------------------
    for (int i = tid; i < NH_DIM; i += 256) sh[i] = h0buf[rp * NH_DIM + i];
    __syncthreads();

    float acc = 0.0f;
#pragma unroll 4
    for (int it = 0; it < NH_DIM / 64; ++it) {
      const int k = it * 64 + gl * 8;
      uint4 u = *reinterpret_cast<const uint4*>(w0 + k);  // 8 bf16
      const float* hv = sh + k;
      acc = fmaf(__uint_as_float(u.x << 16), hv[0], acc);
      acc = fmaf(__uint_as_float(u.x & 0xffff0000u), hv[1], acc);
      acc = fmaf(__uint_as_float(u.y << 16), hv[2], acc);
      acc = fmaf(__uint_as_float(u.y & 0xffff0000u), hv[3], acc);
      acc = fmaf(__uint_as_float(u.z << 16), hv[4], acc);
      acc = fmaf(__uint_as_float(u.z & 0xffff0000u), hv[5], acc);
      acc = fmaf(__uint_as_float(u.w << 16), hv[6], acc);
      acc = fmaf(__uint_as_float(u.w & 0xffff0000u), hv[7], acc);
    }
    acc += __shfl_xor(acc, 1);
    acc += __shfl_xor(acc, 2);
    acc += __shfl_xor(acc, 4);
    acc += Z0[(size_t)t * G4 + row];

    {
      const float gi = __shfl(acc, 0), gf = __shfl(acc, 8);
      const float gg = __shfl(acc, 16), go = __shfl(acc, 24);
      const float iv = sigmoidf_(gi), fv = sigmoidf_(gf);
      const float gv = tanhf(gg),     ov = sigmoidf_(go);
      c0 = fmaf(fv, c0, iv * gv);
      const float h0v = ov * tanhf(c0);
      if (lane == 0) h0buf[p * NH_DIM + cell] = h0v;
    }

    // ---- device-wide barrier (one per timestep) ---------------------------
    __threadfence();
    __syncthreads();
    if (tid == 0) {
      const unsigned g = *vgen;
      if (atomicAdd(syncp, 1u) == (unsigned)(REC_WGS - 1)) {
        *(volatile unsigned*)syncp = 0u;   // reset count before release
        __threadfence();
        atomicAdd(syncp + 1, 1u);          // release generation
      } else {
        while (*vgen == g) __builtin_amdgcn_s_sleep(2);
      }
    }
    __syncthreads();
    __threadfence();

    // ---- layer 1: gates = b1 + [w_ih1|w_hh1] @ [h0_new; h1_prev] ----------
    for (int i = tid; i < NH_DIM; i += 256) {
      sh[i]          = h0buf[p * NH_DIM + i];
      sh[NH_DIM + i] = h1buf[rp * NH_DIM + i];
    }
    __syncthreads();

    acc = 0.0f;
#pragma unroll 4
    for (int it = 0; it < (2 * NH_DIM) / 64; ++it) {
      const int k = it * 64 + gl * 8;
      uint4 u = *reinterpret_cast<const uint4*>(w1 + k);
      const float* hv = sh + k;
      acc = fmaf(__uint_as_float(u.x << 16), hv[0], acc);
      acc = fmaf(__uint_as_float(u.x & 0xffff0000u), hv[1], acc);
      acc = fmaf(__uint_as_float(u.y << 16), hv[2], acc);
      acc = fmaf(__uint_as_float(u.y & 0xffff0000u), hv[3], acc);
      acc = fmaf(__uint_as_float(u.z << 16), hv[4], acc);
      acc = fmaf(__uint_as_float(u.z & 0xffff0000u), hv[5], acc);
      acc = fmaf(__uint_as_float(u.w << 16), hv[6], acc);
      acc = fmaf(__uint_as_float(u.w & 0xffff0000u), hv[7], acc);
    }
    acc += __shfl_xor(acc, 1);
    acc += __shfl_xor(acc, 2);
    acc += __shfl_xor(acc, 4);
    acc += b1[row];

    {
      const float gi = __shfl(acc, 0), gf = __shfl(acc, 8);
      const float gg = __shfl(acc, 16), go = __shfl(acc, 24);
      const float iv = sigmoidf_(gi), fv = sigmoidf_(gf);
      const float gv = tanhf(gg),     ov = sigmoidf_(go);
      c1 = fmaf(fv, c1, iv * gv);
      const float h1v = ov * tanhf(c1);
      if (lane == 0) {
        h1buf[p * NH_DIM + cell] = h1v;
        hs[(size_t)t * NH_DIM + cell] = h1v;
      }
    }
    __syncthreads();  // protect LDS reuse next iteration
  }
}

// ---------------------------------------------------------------------------
// In-place row softmax over NO_DIM columns (one WG per row, rows = T)
// ---------------------------------------------------------------------------
__global__ void softmax_rows(float* __restrict__ logits) {
  __shared__ float red[256];
  float* row = logits + (size_t)blockIdx.x * NO_DIM;
  const int tid = threadIdx.x;

  float mx = -INFINITY;
  for (int i = tid; i < NO_DIM; i += 256) mx = fmaxf(mx, row[i]);
  red[tid] = mx;
  __syncthreads();
  for (int s = 128; s > 0; s >>= 1) {
    if (tid < s) red[tid] = fmaxf(red[tid], red[tid + s]);
    __syncthreads();
  }
  mx = red[0];
  __syncthreads();

  float sum = 0.0f;
  for (int i = tid; i < NO_DIM; i += 256) {
    const float e = __expf(row[i] - mx);
    row[i] = e;
    sum += e;
  }
  red[tid] = sum;
  __syncthreads();
  for (int s = 128; s > 0; s >>= 1) {
    if (tid < s) red[tid] += red[tid + s];
    __syncthreads();
  }
  const float inv = 1.0f / red[0];
  for (int i = tid; i < NO_DIM; i += 256) row[i] *= inv;
}

// ---------------------------------------------------------------------------
// Value head: out[t] = dot(hs[t], w_val) + b_val   (one wave per row)
// ---------------------------------------------------------------------------
__global__ void value_head(const float* __restrict__ hs,
                           const float* __restrict__ wval,
                           const float* __restrict__ bval,
                           float* __restrict__ out) {
  const int t = blockIdx.x * (blockDim.x >> 5) + (threadIdx.x >> 5);
  if (t >= T_STEPS) return;
  const int lane = threadIdx.x & 31;
  const float* h = hs + (size_t)t * NH_DIM;
  float acc = 0.0f;
  for (int i = lane; i < NH_DIM; i += 32) acc = fmaf(h[i], wval[i], acc);
  for (int m = 16; m > 0; m >>= 1) acc += __shfl_xor(acc, m);
  if (lane == 0) out[t] = acc + bval[0];
}

// ---------------------------------------------------------------------------
// Host-side orchestration
// ---------------------------------------------------------------------------
extern "C" void kernel_launch(void* const* d_in, const int* in_sizes, int n_in,
                              void* d_out, int out_size, void* d_ws, size_t ws_size,
                              hipStream_t stream) {
  (void)in_sizes; (void)n_in; (void)out_size; (void)ws_size;
  const float* x     = (const float*)d_in[0];
  const float* wih0  = (const float*)d_in[1];
  const float* whh0  = (const float*)d_in[2];
  const float* bih0  = (const float*)d_in[3];
  const float* bhh0  = (const float*)d_in[4];
  const float* wih1  = (const float*)d_in[5];
  const float* whh1  = (const float*)d_in[6];
  const float* bih1  = (const float*)d_in[7];
  const float* bhh1  = (const float*)d_in[8];
  const float* wpol  = (const float*)d_in[9];
  const float* bpol  = (const float*)d_in[10];
  const float* wval  = (const float*)d_in[11];
  const float* bval  = (const float*)d_in[12];
  float* out = (float*)d_out;

  // Workspace carve-out (256B aligned chunks)
  size_t off = 0;
  auto carve = [&](size_t bytes) -> char* {
    char* p = (char*)d_ws + off;
    off += (bytes + 255) & ~(size_t)255;
    return p;
  };
  unsigned*       syncp  = (unsigned*)      carve(256);
  unsigned short* Xbf    = (unsigned short*)carve((size_t)T_STEPS * NS_DIM * 2);
  unsigned short* Wih0bf = (unsigned short*)carve((size_t)G4 * NS_DIM * 2);
  unsigned short* Whh0bf = (unsigned short*)carve((size_t)G4 * NH_DIM * 2);
  unsigned short* W1cat  = (unsigned short*)carve((size_t)G4 * 2 * NH_DIM * 2);
  unsigned short* Hsbf   = (unsigned short*)carve((size_t)T_STEPS * NH_DIM * 2);
  unsigned short* Wpolbf = (unsigned short*)carve((size_t)NO_DIM * NH_DIM * 2);
  float* b0    = (float*)carve((size_t)G4 * 4);
  float* b1    = (float*)carve((size_t)G4 * 4);
  float* h0buf = (float*)carve((size_t)2 * NH_DIM * 4);
  float* h1buf = (float*)carve((size_t)2 * NH_DIM * 4);
  float* Z0    = (float*)carve((size_t)T_STEPS * G4 * 4);
  float* hs    = (float*)carve((size_t)T_STEPS * NH_DIM * 4);

  const int BT = 256;
  auto blocks = [](size_t n, int bt) { return (unsigned)((n + bt - 1) / bt); };

  // 0) per-launch init (deterministic: resets barrier state + h buffers)
  init_kernel<<<blocks(G4, BT), BT, 0, stream>>>(bih0, bhh0, bih1, bhh1,
                                                 b0, b1, h0buf, h1buf, syncp);
  // 1) conversions to bf16
  conv_bf16_kernel<<<blocks((size_t)T_STEPS * NS_DIM, BT), BT, 0, stream>>>(
      x, Xbf, T_STEPS * NS_DIM);
  conv_bf16_kernel<<<blocks((size_t)G4 * NS_DIM, BT), BT, 0, stream>>>(
      wih0, Wih0bf, G4 * NS_DIM);
  conv_bf16_kernel<<<blocks((size_t)G4 * NH_DIM, BT), BT, 0, stream>>>(
      whh0, Whh0bf, G4 * NH_DIM);
  concat_w1_kernel<<<blocks((size_t)G4 * 2 * NH_DIM, BT), BT, 0, stream>>>(
      wih1, whh1, W1cat);
  conv_bf16_kernel<<<blocks((size_t)NO_DIM * NH_DIM, BT), BT, 0, stream>>>(
      wpol, Wpolbf, NO_DIM * NH_DIM);

  // 2) Z0 = X @ W_ih0^T + b0   (M=8192, N=4096, K=512) — WMMA
  {
    const unsigned tiles = (T_STEPS / 16) * (G4 / 16);
    wmma_gemm_bf16<<<tiles / 8, 256, 0, stream>>>(Xbf, Wih0bf, b0, Z0,
                                                  T_STEPS, G4, NS_DIM);
  }

  // 3) sequential recurrence (persistent, one grid barrier per timestep)
  lstm_persistent<<<REC_WGS, 256, 0, stream>>>(Whh0bf, W1cat, Z0, b1,
                                               h0buf, h1buf, hs, syncp);

  // 4) policy logits = hs @ w_pol^T + b_pol   (M=8192, N=512, K=1024) — WMMA
  conv_bf16_kernel<<<blocks((size_t)T_STEPS * NH_DIM, BT), BT, 0, stream>>>(
      hs, Hsbf, T_STEPS * NH_DIM);
  {
    const unsigned tiles = (T_STEPS / 16) * (NO_DIM / 16);
    wmma_gemm_bf16<<<tiles / 8, 256, 0, stream>>>(Hsbf, Wpolbf, bpol, out,
                                                  T_STEPS, NO_DIM, NH_DIM);
  }

  // 5) softmax in place on the policy region of d_out
  softmax_rows<<<T_STEPS, 256, 0, stream>>>(out);

  // 6) value head -> tail of d_out
  value_head<<<T_STEPS / 8, 256, 0, stream>>>(hs, wval, bval,
                                              out + (size_t)T_STEPS * NO_DIM);
}
